// EMAModule_33200097198792
// MI455X (gfx1250) — matrix operand order repaired and checked by reading, not compiled, not run on hardware
//
#include <hip/hip_runtime.h>
#include <math.h>

// ---------------- problem constants (from reference) ----------------
#define BB 16
#define CC 512
#define NN 4096   // 64*64
#define KK 64
#define NSTAGES 3
#define EPSF 1e-5f

typedef __attribute__((ext_vector_type(16))) __bf16 v16bf;
typedef __attribute__((ext_vector_type(8)))  __bf16 v8bf;
typedef __attribute__((ext_vector_type(2)))  __bf16 v2bf;
typedef __attribute__((ext_vector_type(8)))  float  v8f;

static __device__ __forceinline__ v16bf cat16(v8bf lo, v8bf hi) {
    return __builtin_shufflevector(lo, hi, 0,1,2,3,4,5,6,7,8,9,10,11,12,13,14,15);
}
static __device__ __forceinline__ v8f vzero8() {
    v8f z;
#pragma unroll
    for (int i = 0; i < 8; ++i) z[i] = 0.f;
    return z;
}
static __device__ __forceinline__ v8f wmma_bf16(v16bf a, v16bf b, v8f c) {
    // D = A(16x32 bf16) x B(32x16 bf16) + C(16x16 f32)
    return __builtin_amdgcn_wmma_f32_16x16x32_bf16(false, a, false, b, (short)0, c,
                                                   false, false);
}

// Async global->LDS copy of one 16B chunk per lane (CDNA5 direct path,
// tracked by ASYNCcnt; bypasses VGPRs). ldsoff = LDS byte offset (= low 32
// bits of the flat shared address), gaddr = 64-bit global address.
static __device__ __forceinline__ void async_g2l_b128(unsigned ldsoff,
                                                      unsigned long long gaddr) {
    asm volatile("global_load_async_to_lds_b128 %0, %1, off"
                 :: "v"(ldsoff), "v"(gaddr) : "memory");
}
static __device__ __forceinline__ void wait_async0() {
    asm volatile("s_wait_asynccnt 0x0" ::: "memory");
}

// ---------------- prep: f32 feats -> bf16 in two layouts ----------------
// f_b  : [B][C][N]  (M-step B operand)
// fT_b : [B][N][C]  (E-step A operand)
__global__ void ema_prep_f(const float* __restrict__ feats,
                           __bf16* __restrict__ f_b,
                           __bf16* __restrict__ fT_b) {
    __shared__ float tile[32][33];
    const int n0 = blockIdx.x * 32, c0 = blockIdx.y * 32, b = blockIdx.z;
    const int t = threadIdx.x, col = t & 31, r = t >> 5;   // r in 0..7
#pragma unroll
    for (int i = 0; i < 4; ++i) {
        const int cr = r + i * 8;
        const float v = feats[((size_t)(b * CC + c0 + cr)) * NN + n0 + col];
        tile[cr][col] = v;
        f_b[((size_t)(b * CC + c0 + cr)) * NN + n0 + col] = (__bf16)v;
    }
    __syncthreads();
#pragma unroll
    for (int i = 0; i < 4; ++i) {
        const int nr = r + i * 8;
        fT_b[((size_t)(b * NN + n0 + nr)) * CC + c0 + col] = (__bf16)tile[col][nr];
    }
}

// baseT[b][k][c] = bases[c][k] broadcast over b  (E-step B operand)
__global__ void ema_prep_base(const float* __restrict__ bases,
                              __bf16* __restrict__ baseT) {
    const int idx = blockIdx.x * 256 + threadIdx.x;   // = ((b*KK + k)*CC + c)
    if (idx >= BB * KK * CC) return;
    const int c = idx & (CC - 1);
    const int k = (idx >> 9) & (KK - 1);
    baseT[idx] = (__bf16)bases[c * KK + k];
}

// ---------------- E-step: att = softmax_k( fT x baseT^T ) ----------------
// grid (N/128, B), 256 thr (8 waves). Wave w owns rows n0+16w..+15, all K=64.
__global__ void ema_estep(const __bf16* __restrict__ fT,
                          const __bf16* __restrict__ baseT,
                          __bf16* __restrict__ att_nk,   // [B][N][K]
                          __bf16* __restrict__ att_kn) { // [B][K][N]
    extern __shared__ char smem[];
    __bf16* sB = (__bf16*)smem;                     // baseT[b]: 64*512 bf16 = 64 KiB
    float*  sS = (float*)(smem + KK * CC * 2);      // 8 waves * 16 * 68 f32

    const int b = blockIdx.y, n0 = blockIdx.x * 128;
    const int t = threadIdx.x, wave = t >> 5, lane = t & 31;
    const int m = lane & 15, h = lane >> 4;

    {   // async stage of baseT[b] into LDS (16B chunk per lane per iter)
        const __bf16* src = baseT + (size_t)b * KK * CC;
#pragma unroll
        for (int i = 0; i < 16; ++i) {
            const int j = t + i * 256;                       // chunk id < 4096
            async_g2l_b128((unsigned)(uintptr_t)(sB + j * 8),
                           (unsigned long long)(uintptr_t)(src + j * 8));
        }
        wait_async0();
    }
    __syncthreads();

    v8f acc[4];
#pragma unroll
    for (int kt = 0; kt < 4; ++kt) acc[kt] = vzero8();

    const __bf16* Arow = fT + ((size_t)(b * NN + n0 + wave * 16 + m)) * CC;
    v8bf alo = *(const v8bf*)(Arow + 8 * h);
    v8bf ahi = *(const v8bf*)(Arow + 16 + 8 * h);
    for (int cs = 0; cs < CC; cs += 32) {
        const v16bf a = cat16(alo, ahi);
        if (cs + 32 < CC) {   // double-buffer the streamed A operand
            __builtin_prefetch(Arow + cs + 160, 0, 1);       // global_prefetch_b8
            alo = *(const v8bf*)(Arow + cs + 32 + 8 * h);
            ahi = *(const v8bf*)(Arow + cs + 48 + 8 * h);
        }
        v16bf bb[4];
#pragma unroll
        for (int kt = 0; kt < 4; ++kt) {
            const __bf16* Bp = sB + (kt * 16 + m) * CC + cs + 8 * h;
            bb[kt] = cat16(*(const v8bf*)Bp, *(const v8bf*)(Bp + 16));
        }
#pragma unroll
        for (int kt = 0; kt < 4; ++kt) acc[kt] = wmma_bf16(a, bb[kt], acc[kt]);
    }

    // spill D tiles to per-wave LDS scratch: rows n_local 0..15 x 64 k
    float* S = sS + wave * 16 * 68;
#pragma unroll
    for (int kt = 0; kt < 4; ++kt)
#pragma unroll
        for (int r = 0; r < 8; ++r)
            S[(r + 8 * h) * 68 + kt * 16 + m] = acc[kt][r];
    __syncthreads();

    // softmax over k: all 32 lanes active; row = lane&15, half h does 32 of 64 k
    {
        float* R = S + m * 68 + h * 32;
        float mx = -3.4e38f;
#pragma unroll
        for (int k = 0; k < 32; ++k) mx = fmaxf(mx, R[k]);
        mx = fmaxf(mx, __shfl_xor(mx, 16, 32));
        float sum = 0.f;
#pragma unroll
        for (int k = 0; k < 32; ++k) { const float e = __expf(R[k] - mx); R[k] = e; sum += e; }
        sum += __shfl_xor(sum, 16, 32);
        const float inv = 1.f / sum;
#pragma unroll
        for (int k = 0; k < 32; ++k) R[k] *= inv;   // normalized att back into LDS
    }
    __syncthreads();

    // cooperative coalesced stores of both layouts (packed 2xbf16 dwords)
    for (int j = t; j < 128 * 32; j += 256) {        // att_nk: nl = j>>5, kpair = j&31
        const int nl = j >> 5, k2 = (j & 31) * 2;
        const float* Sp = sS + (nl >> 4) * (16 * 68) + (nl & 15) * 68;
        v2bf pr; pr[0] = (__bf16)Sp[k2]; pr[1] = (__bf16)Sp[k2 + 1];
        *(v2bf*)(att_nk + ((size_t)(b * NN + n0 + nl)) * KK + k2) = pr;
    }
    for (int j = t; j < 64 * 64; j += 256) {         // att_kn: k = j>>6, npair = j&63
        const int k = j >> 6, p2 = (j & 63) * 2;
        v2bf pr;
        pr[0] = (__bf16)sS[((p2)     >> 4) * (16 * 68) + ((p2)     & 15) * 68 + k];
        pr[1] = (__bf16)sS[((p2 + 1) >> 4) * (16 * 68) + ((p2 + 1) & 15) * 68 + k];
        *(v2bf*)(att_kn + ((size_t)(b * KK + k)) * NN + n0 + p2) = pr;
    }
}

// ---------------- M-step GEMM: rawT[k][c] = sum_n att_kn[k][n] * f[c][n] ----
// grid (4 n-chunks, 4 c-chunks, B), 256 thr. Each workgroup async-stages a
// 64x1024 bf16 chunk of att_kn in LDS (128 KiB) and writes its own rawT slice.
__global__ void ema_mstep(const __bf16* __restrict__ f_b,
                          const __bf16* __restrict__ att_kn,
                          float* __restrict__ rawT4) {   // [4][B][K][C]
    extern __shared__ char smem[];
    __bf16* sA = (__bf16*)smem;                          // [64][1024]
    const int nch = blockIdx.x, cch = blockIdx.y, b = blockIdx.z;
    const int n0 = nch * 1024, c0 = cch * 128;
    const int t = threadIdx.x, wave = t >> 5, lane = t & 31;
    const int m = lane & 15, h = lane >> 4;

    {   // async stage att_kn[b][0:64][n0:n0+1024] into LDS
        const __bf16* src = att_kn + (size_t)b * KK * NN;
#pragma unroll
        for (int i = 0; i < 32; ++i) {
            const int j = t + i * 256;                 // chunk id < 8192
            const int k = j >> 7, off = (j & 127) * 8;
            async_g2l_b128((unsigned)(uintptr_t)(sA + k * 1024 + off),
                           (unsigned long long)(uintptr_t)(src + (size_t)k * NN + n0 + off));
        }
        wait_async0();
    }
    __syncthreads();

    const int c = c0 + wave * 16 + m;
    const __bf16* Bcol = f_b + ((size_t)(b * CC + c)) * NN + n0;
    v8f acc[4];
#pragma unroll
    for (int mt = 0; mt < 4; ++mt) acc[mt] = vzero8();

    v8bf blo = *(const v8bf*)(Bcol + 16 * h);
    v8bf bhi = *(const v8bf*)(Bcol + 16 * h + 8);
    for (int ns = 0; ns < 1024; ns += 32) {
        const v16bf bb = cat16(blo, bhi);
        if (ns + 32 < 1024) {   // double-buffer the streamed B operand
            __builtin_prefetch(Bcol + ns + 160, 0, 1);
            blo = *(const v8bf*)(Bcol + ns + 32 + 16 * h);
            bhi = *(const v8bf*)(Bcol + ns + 32 + 16 * h + 8);
        }
        v16bf av[4];
#pragma unroll
        for (int mt = 0; mt < 4; ++mt) {
            const __bf16* Ap = sA + (mt * 16 + m) * 1024 + ns + 16 * h;
            av[mt] = cat16(*(const v8bf*)Ap, *(const v8bf*)(Ap + 8));
        }
#pragma unroll
        for (int mt = 0; mt < 4; ++mt) acc[mt] = wmma_bf16(av[mt], bb, acc[mt]);
    }

    float* out = rawT4 + ((size_t)nch * BB + b) * KK * CC;
#pragma unroll
    for (int mt = 0; mt < 4; ++mt)
#pragma unroll
        for (int r = 0; r < 8; ++r)
            out[((size_t)(mt * 16 + r + 8 * h)) * CC + c] = acc[mt][r];
}

// ---------------- normalize: fold l1(att) + l2(base) into one scale --------
// base_col = raw_col / (eps*(eps + colsum_k) + ||raw_col||)  [algebraically
// identical to l1-normalizing att then l2-normalizing the M-step output]
__global__ void ema_norm(const float* __restrict__ rawT4,
                         const __bf16* __restrict__ att_kn,
                         __bf16* __restrict__ baseT,    // [B][K][C]
                         __bf16* __restrict__ baseb) {  // [B][C][K]
    __shared__ float red[256], red2[256];
    const int k = blockIdx.x, b = blockIdx.y, t = threadIdx.x;

    // deterministic colsum_k = sum_n att[b][n][k] (att_kn row is contiguous)
    const __bf16* ar = att_kn + ((size_t)(b * KK + k)) * NN;
    float cs = 0.f;
    for (int i = t; i < NN; i += 256) cs += (float)ar[i];
    red2[t] = cs;

    // combine the 4 n-chunk slices of rawT
    const size_t roff = ((size_t)(b * KK + k)) * CC;
    float v0 = 0.f, v1 = 0.f;
#pragma unroll
    for (int s4 = 0; s4 < 4; ++s4) {
        const float* rp = rawT4 + (size_t)s4 * BB * KK * CC + roff;
        v0 += rp[t];
        v1 += rp[t + 256];
    }
    red[t] = v0 * v0 + v1 * v1;
    __syncthreads();
    for (int off = 128; off > 0; off >>= 1) {
        if (t < off) { red[t] += red[t + off]; red2[t] += red2[t + off]; }
        __syncthreads();
    }
    const float scale = 1.f / (EPSF * (EPSF + red2[0]) + sqrtf(red[0]));
    const float a0 = v0 * scale, a1 = v1 * scale;
    baseT[roff + t]       = (__bf16)a0;
    baseT[roff + t + 256] = (__bf16)a1;
    baseb[((size_t)(b * CC + t)) * KK + k]       = (__bf16)a0;
    baseb[((size_t)(b * CC + t + 256)) * KK + k] = (__bf16)a1;
}

// ---------------- recon: out[c][n] = sum_k base[c][k] * att[n][k] ----------
// grid (N/256, C/32, B), 256 thr; wave -> (cw 0..1, nw 0..3): 16c x 64n tiles.
__global__ void ema_recon(const __bf16* __restrict__ baseb,
                          const __bf16* __restrict__ att_nk,
                          float* __restrict__ out) {
    const int b = blockIdx.z, cblk = blockIdx.y, nblk = blockIdx.x;
    const int t = threadIdx.x, wave = t >> 5, lane = t & 31;
    const int m = lane & 15, h = lane >> 4;
    const int ctile = cblk * 32 + (wave >> 2) * 16;
    const int nbase = nblk * 256 + (wave & 3) * 64;

    const __bf16* Arow = baseb + ((size_t)(b * CC + ctile + m)) * KK;
    v8f acc[4];
#pragma unroll
    for (int j = 0; j < 4; ++j) acc[j] = vzero8();

#pragma unroll
    for (int ks = 0; ks < KK; ks += 32) {
        const v16bf a = cat16(*(const v8bf*)(Arow + ks + 8 * h),
                              *(const v8bf*)(Arow + ks + 16 + 8 * h));
        v16bf bv[4];
#pragma unroll
        for (int j = 0; j < 4; ++j) {
            const int n = nbase + j * 16 + m;
            const __bf16* Bp = att_nk + ((size_t)(b * NN + n)) * KK + ks + 8 * h;
            bv[j] = cat16(*(const v8bf*)Bp, *(const v8bf*)(Bp + 16));
        }
#pragma unroll
        for (int j = 0; j < 4; ++j) acc[j] = wmma_bf16(a, bv[j], acc[j]);
    }
#pragma unroll
    for (int j = 0; j < 4; ++j)
#pragma unroll
        for (int r = 0; r < 8; ++r)
            out[((size_t)(b * CC + ctile + r + 8 * h)) * NN + nbase + j * 16 + m] =
                acc[j][r];
}

// ---------------- launch ----------------
extern "C" void kernel_launch(void* const* d_in, const int* in_sizes, int n_in,
                              void* d_out, int out_size, void* d_ws, size_t ws_size,
                              hipStream_t stream) {
    const float* feats = (const float*)d_in[0];   // [16,512,64,64] f32
    const float* bases = (const float*)d_in[1];   // [1,512,64] f32
    float* out = (float*)d_out;                   // [16,512,64,64] f32
    char* ws = (char*)d_ws;

    __bf16* f_b    = (__bf16*)(ws);                 // 64 MiB   [B][C][N]
    __bf16* fT_b   = (__bf16*)(ws + 67108864ull);   // 64 MiB   [B][N][C]
    __bf16* att_nk = (__bf16*)(ws + 134217728ull);  //  8 MiB   [B][N][K]
    __bf16* att_kn = (__bf16*)(ws + 142606336ull);  //  8 MiB   [B][K][N]
    __bf16* baseT  = (__bf16*)(ws + 150994944ull);  //  1 MiB   [B][K][C]
    __bf16* baseb  = (__bf16*)(ws + 152043520ull);  //  1 MiB   [B][C][K]
    float*  rawT4  = (float*)(ws + 153092096ull);   //  8 MiB   [4][B][K][C]

    ema_prep_f<<<dim3(NN / 32, CC / 32, BB), 256, 0, stream>>>(feats, f_b, fT_b);
    ema_prep_base<<<(BB * KK * CC) / 256, 256, 0, stream>>>(bases, baseT);

    const size_t smemE = (size_t)KK * CC * 2 + 8 * 16 * 68 * 4;  // ~100 KiB
    const size_t smemM = (size_t)KK * 1024 * 2;                  // 128 KiB
    for (int s = 0; s < NSTAGES; ++s) {
        ema_estep<<<dim3(NN / 128, BB), 256, smemE, stream>>>(fT_b, baseT, att_nk, att_kn);
        ema_mstep<<<dim3(4, 4, BB), 256, smemM, stream>>>(f_b, att_kn, rawT4);
        ema_norm<<<dim3(KK, BB), 256, 0, stream>>>(rawT4, att_kn, baseT, baseb);
    }
    ema_recon<<<dim3(NN / 256, CC / 32, BB), 256, 0, stream>>>(baseb, att_nk, out);
}
